// MyRNN_15616501088566
// MI455X (gfx1250) — matrix operand order repaired and checked by reading, not compiled, not run on hardware
//
#include <hip/hip_runtime.h>
#include <hip/hip_bf16.h>

typedef __attribute__((ext_vector_type(16))) _Float16 v16h;
typedef __attribute__((ext_vector_type(8)))  _Float16 v8h;
typedef __attribute__((ext_vector_type(8)))  float    v8f;

#define BATCH 64
#define SEQ   80
#define VOCAB 10000
#define EMBED 100
#define UNITS 2048

// WGP-scope prefetch (scope 0): pulls the line into ALL cache levels,
// including the WGP-local cache. clang's __builtin_prefetch only reaches
// SE/SYS scope, which per ISA 10.5 skips the WGP cache entirely — useless
// when the data is already L2-resident. No "memory" clobber: prefetch has
// no architectural effect, so the scheduler stays free to reorder loads.
__device__ __forceinline__ void prefetch_wgp(const void* p) {
  asm volatile("global_prefetch_b8 %0, off" :: "v"(p));
}

// ---------------------------------------------------------------------------
// Prep: W[K][N] fp32 -> Wt[N][K] f16 (transpose + convert) so the WMMA
// B-operand (per-lane contiguous K) is a contiguous 32B vector load.
// ---------------------------------------------------------------------------
__global__ void w_transpose_f16(const float* __restrict__ W,
                                _Float16* __restrict__ Wt, int K, int N) {
  int idx = blockIdx.x * blockDim.x + threadIdx.x;
  if (idx >= K * N) return;
  int n = idx / K;
  int k = idx - n * K;
  Wt[idx] = (_Float16)W[(long)k * N + n];
}

// ---------------------------------------------------------------------------
// Prep: fused embedding gather + xt @ Wx0 for ALL timesteps.
// XW0[t][b][u] = sum_e emb[inputs[b][t]][e] * Wx0[e][u]   (f16 out)
// ---------------------------------------------------------------------------
__global__ void embed_xw0(const int* __restrict__ inputs,
                          const float* __restrict__ emb,
                          const float* __restrict__ Wx0,
                          _Float16* __restrict__ XW0) {
  long idx = (long)blockIdx.x * blockDim.x + threadIdx.x;
  const long total = (long)SEQ * BATCH * UNITS;
  if (idx >= total) return;
  int  u  = (int)(idx % UNITS);
  long tb = idx / UNITS;
  int  b  = (int)(tb % BATCH);
  int  t  = (int)(tb / BATCH);
  int tok = inputs[b * SEQ + t];
  const float* e = emb + (long)tok * EMBED;
  float acc = 0.f;
  #pragma unroll 4
  for (int i = 0; i < EMBED; ++i) acc += e[i] * Wx0[(long)i * UNITS + u];
  XW0[idx] = (_Float16)acc;
}

__global__ void zero_f16(_Float16* __restrict__ p, int n) {
  int i = blockIdx.x * blockDim.x + threadIdx.x;
  if (i < n) p[i] = (_Float16)0.f;
}

// ---------------------------------------------------------------------------
// One recurrent step-layer (split-K x4):
//   Hout[64][2048] = tanh(Cinit + bias + A1 @ W1t^T (+ A2 @ W2t^T))
// Grid: 512 blocks (one 16x16 tile each) x 128 threads (4 waves).
// Each wave accumulates one K-chunk with software-pipelined loads,
// partials reduced through LDS, wave 0 runs the tanh epilogue.
//   layer0 (A2==null): K=2048, chunks of 512  (16 WMMA/wave)
//   layer1 (A2!=null): virtual K=4096 -> waves 0-1 on A1/W1t, 2-3 on A2/W2t,
//                      chunks of 1024 (32 WMMA/wave)
// ---------------------------------------------------------------------------
__global__ __launch_bounds__(128) void rnn_step(
    const _Float16* __restrict__ A1, const _Float16* __restrict__ W1t,
    const _Float16* __restrict__ A2, const _Float16* __restrict__ W2t,
    const _Float16* __restrict__ Cinit, const float* __restrict__ bias,
    _Float16* __restrict__ Hout) {
  const int lane = threadIdx.x & 31;
  const int wave = threadIdx.x >> 5;          // 0..3 (split-K id)
  const int tile = blockIdx.x;                // 0..511
  const int m0 = (tile & 3) * 16;             // batch-tile
  const int n0 = (tile >> 2) * 16;            // unit-tile
  const int half = lane >> 4;                 // 0 | 1
  const int l15  = lane & 15;
  const int col  = n0 + l15;                  // B/C/D column for this lane
  const int arow = m0 + l15;                  // A row for this lane
  const int ka = half * 16;                   // A K-phase  (contiguous K, matches B)
  const int kb = half * 16;                   // B K-phase  (contiguous K, matches A)

  // Pick this wave's K-chunk (uniform per wave -> EXEC stays all-ones).
  const _Float16* Asrc;
  const _Float16* Wsrc;
  int k0, klen;
  if (A2) {
    Asrc = (wave < 2) ? A1 : A2;
    Wsrc = (wave < 2) ? W1t : W2t;
    k0   = (wave & 1) * 1024;
    klen = 1024;
  } else {
    Asrc = A1;
    Wsrc = W1t;
    k0   = wave * 512;
    klen = 512;
  }

  const _Float16* ap = Asrc + (long)arow * UNITS + k0;
  const _Float16* bp = Wsrc + (long)col  * UNITS + k0;

  v8f c = {};  // split-K partial; bias/Cinit added after reduction

  // Software-pipelined K-loop: loads for iteration i+1 are issued before the
  // WMMA of iteration i waits, hiding L2 latency (1 wave/SIMD regime).
  v8h alo = *(const v8h*)(ap + ka);
  v8h ahi = *(const v8h*)(ap + 8 + ka);
  v16h b  = *(const v16h*)(bp + kb);
  for (int kk = 32; kk < klen; kk += 32) {
    v8h alo_n = *(const v8h*)(ap + kk + ka);
    v8h ahi_n = *(const v8h*)(ap + kk + 8 + ka);
    v16h b_n  = *(const v16h*)(bp + kk + kb);
    // Pull next weight chunk from L2 into the WGP cache ahead of the loads.
    prefetch_wgp((const void*)(bp + kk + 256));
    v16h a;
    #pragma unroll
    for (int i = 0; i < 8; ++i) { a[i] = alo[i]; a[i + 8] = ahi[i]; }
    c = __builtin_amdgcn_wmma_f32_16x16x32_f16(false, a, false, b,
                                               (short)0, c, false, false);
    alo = alo_n; ahi = ahi_n; b = b_n;
  }
  {
    v16h a;
    #pragma unroll
    for (int i = 0; i < 8; ++i) { a[i] = alo[i]; a[i + 8] = ahi[i]; }
    c = __builtin_amdgcn_wmma_f32_16x16x32_f16(false, a, false, b,
                                               (short)0, c, false, false);
  }

  // Split-K reduction through LDS (conflict-free: lane is fastest index).
  __shared__ float red[4][8][32];
  if (wave != 0) {
    #pragma unroll
    for (int i = 0; i < 8; ++i) red[wave][i][lane] = c[i];
  }
  __syncthreads();
  if (wave == 0) {
    const float bv = bias[col];
    #pragma unroll
    for (int i = 0; i < 8; ++i) {
      int r = m0 + half * 8 + i;              // C/D row layout
      float v = c[i] + red[1][i][lane] + red[2][i][lane] + red[3][i][lane] + bv;
      if (Cinit) v += (float)Cinit[(long)r * UNITS + col];
      Hout[(long)r * UNITS + col] = (_Float16)tanhf(v);
    }
  }
}

// ---------------------------------------------------------------------------
// Head: out[b] = sigmoid(h1[b] . Wo + bo). One wave32 per batch row.
// ---------------------------------------------------------------------------
__global__ void out_head(const _Float16* __restrict__ h1,
                         const float* __restrict__ Wo,
                         const float* __restrict__ bo,
                         float* __restrict__ out) {
  int b = blockIdx.x;
  int lane = threadIdx.x;
  float acc = 0.f;
  for (int u = lane; u < UNITS; u += 32)
    acc += (float)h1[(long)b * UNITS + u] * Wo[u];
  #pragma unroll
  for (int off = 16; off > 0; off >>= 1)
    acc += __shfl_down(acc, off, 32);
  if (lane == 0) out[b] = 1.f / (1.f + expf(-(acc + bo[0])));
}

// ---------------------------------------------------------------------------
extern "C" void kernel_launch(void* const* d_in, const int* in_sizes, int n_in,
                              void* d_out, int out_size, void* d_ws, size_t ws_size,
                              hipStream_t stream) {
  const int*   inputs = (const int*)  d_in[0];
  const float* emb    = (const float*)d_in[1];
  const float* Wx0    = (const float*)d_in[2];
  const float* Wh0    = (const float*)d_in[3];
  const float* b0     = (const float*)d_in[4];
  const float* Wx1    = (const float*)d_in[5];
  const float* Wh1    = (const float*)d_in[6];
  const float* b1     = (const float*)d_in[7];
  const float* Wo     = (const float*)d_in[8];
  const float* bo     = (const float*)d_in[9];
  float* out = (float*)d_out;

  // Workspace carve-up (~47 MB total).
  char* ws = (char*)d_ws;
  size_t off = 0;
  auto carve = [&](size_t bytes) {
    void* p = ws + off;
    off += (bytes + 255) & ~(size_t)255;
    return p;
  };
  _Float16* XW0  = (_Float16*)carve((size_t)SEQ * BATCH * UNITS * 2);
  _Float16* Wh0t = (_Float16*)carve((size_t)UNITS * UNITS * 2);
  _Float16* Wx1t = (_Float16*)carve((size_t)UNITS * UNITS * 2);
  _Float16* Wh1t = (_Float16*)carve((size_t)UNITS * UNITS * 2);
  _Float16* h0a  = (_Float16*)carve((size_t)BATCH * UNITS * 2);
  _Float16* h0b  = (_Float16*)carve((size_t)BATCH * UNITS * 2);
  _Float16* h1a  = (_Float16*)carve((size_t)BATCH * UNITS * 2);
  _Float16* h1b  = (_Float16*)carve((size_t)BATCH * UNITS * 2);

  // One-time prep (inside capture; deterministic, idempotent).
  const int wn = UNITS * UNITS;
  w_transpose_f16<<<(wn + 255) / 256, 256, 0, stream>>>(Wh0, Wh0t, UNITS, UNITS);
  w_transpose_f16<<<(wn + 255) / 256, 256, 0, stream>>>(Wx1, Wx1t, UNITS, UNITS);
  w_transpose_f16<<<(wn + 255) / 256, 256, 0, stream>>>(Wh1, Wh1t, UNITS, UNITS);

  const long xn = (long)SEQ * BATCH * UNITS;
  embed_xw0<<<(int)((xn + 255) / 256), 256, 0, stream>>>(inputs, emb, Wx0, XW0);

  const int hn = BATCH * UNITS;
  zero_f16<<<(hn + 255) / 256, 256, 0, stream>>>(h0a, hn);
  zero_f16<<<(hn + 255) / 256, 256, 0, stream>>>(h1a, hn);

  // The scan: 2 WMMA-GEMM kernels per timestep, ping-ponged h buffers.
  _Float16 *h0c = h0a, *h0n = h0b, *h1c = h1a, *h1n = h1b;
  for (int t = 0; t < SEQ; ++t) {
    rnn_step<<<512, 128, 0, stream>>>(h0c, Wh0t, nullptr, nullptr,
                                      XW0 + (long)t * BATCH * UNITS, b0, h0n);
    rnn_step<<<512, 128, 0, stream>>>(h0n, Wx1t, h1c, Wh1t,
                                      nullptr, b1, h1n);
    _Float16* tmp;
    tmp = h0c; h0c = h0n; h0n = tmp;
    tmp = h1c; h1c = h1n; h1n = tmp;
  }

  out_head<<<BATCH, 32, 0, stream>>>(h1c, Wo, bo, out);
}
